// GatedLinearAttention_88459146428925
// MI455X (gfx1250) — matrix operand order repaired and checked by reading, not compiled, not run on hardware
//
#include <hip/hip_runtime.h>
#include <math.h>

// ---------------------------------------------------------------------------
// Types for CDNA5 WMMA (gfx1250, wave32)
// ---------------------------------------------------------------------------
typedef __attribute__((ext_vector_type(16))) __bf16 bf16x16;
typedef __attribute__((ext_vector_type(8)))  float  floatx8;
typedef __attribute__((ext_vector_type(4)))  int    v4i;

union FragBF { unsigned int u[8]; bf16x16 v; };

__device__ __forceinline__ unsigned short f2bf(float a) {
    unsigned int ua = __float_as_uint(a);
    return (unsigned short)((ua + 0x7FFFu + ((ua >> 16) & 1u)) >> 16);  // RNE
}
__device__ __forceinline__ unsigned int pack2bf(float a, float b) {
    return (unsigned int)f2bf(a) | ((unsigned int)f2bf(b) << 16);
}

// CDNA5 async global->LDS copy (ASYNCcnt path), with portable fallback.
#if defined(__AMDGCN__) && __has_builtin(__builtin_amdgcn_global_load_async_to_lds_b128)
#define GLA_ASYNC 1
typedef __attribute__((address_space(1))) v4i gv4i;
typedef __attribute__((address_space(3))) v4i lv4i;
__device__ __forceinline__ void async_cp16(const unsigned short* g, unsigned short* l) {
    __builtin_amdgcn_global_load_async_to_lds_b128((gv4i*)g, (lv4i*)l, 0, 0);
}
#if __has_builtin(__builtin_amdgcn_s_wait_asynccnt)
#define WAIT_ASYNC(n) __builtin_amdgcn_s_wait_asynccnt(n)
#else
#define WAIT_ASYNC(n) asm volatile("s_wait_asynccnt %0" ::"n"(n))
#endif
#else
#define GLA_ASYNC 0
__device__ __forceinline__ void async_cp16(const unsigned short* g, unsigned short* l) {
    *(uint4*)l = *(const uint4*)g;   // sync fallback: vmem load + ds store
}
#define WAIT_ASYNC(n)
#endif

// ---------------------------------------------------------------------------
// f32 -> bf16 elementwise (n4 = count/4)
// ---------------------------------------------------------------------------
__launch_bounds__(256)
__global__ void conv_bf16(const float* __restrict__ src, unsigned short* __restrict__ dst, int n4)
{
    int i = blockIdx.x * 256 + threadIdx.x;
    if (i >= n4) return;
    float4 f = *(const float4*)(src + (size_t)i * 4);
    uint2 u;
    u.x = pack2bf(f.x, f.y);
    u.y = pack2bf(f.z, f.w);
    *(uint2*)(dst + (size_t)i * 4) = u;
}

// ---------------------------------------------------------------------------
// f32 (KxN) -> bf16 transposed (NxK), 32x32 tiles through LDS.
// ---------------------------------------------------------------------------
__launch_bounds__(256)
__global__ void convT_bf16(const float* __restrict__ W, unsigned short* __restrict__ Wt,
                           int K, int N)
{
    __shared__ unsigned short t[32][33];
    int n0 = blockIdx.x * 32, k0 = blockIdx.y * 32;
    int tx = threadIdx.x & 31, ty = threadIdx.x >> 5;   // ty 0..7
#pragma unroll
    for (int r = 0; r < 4; ++r) {
        int kk = ty + r * 8;
        t[kk][tx] = f2bf(W[(size_t)(k0 + kk) * N + n0 + tx]);
    }
    __syncthreads();
#pragma unroll
    for (int r = 0; r < 4; ++r) {
        int nn = ty + r * 8;
        Wt[(size_t)(n0 + nn) * K + k0 + tx] = t[tx][nn];
    }
}

// ---------------------------------------------------------------------------
// GEMM: C(MxN) = alpha * A @ B + bias[n], where A is bf16 MxK row-major and
// Bt is bf16 NxK row-major (i.e. B pre-transposed). Double-buffered LDS with
// async global->LDS staging. Block tile 128x128; 8 waves x (32x64) each.
// Requires M%128==0, N%128==0, K%32==0.
// ---------------------------------------------------------------------------
__launch_bounds__(256)
__global__ void gemm_bf16bf16(const unsigned short* __restrict__ A,
                              const unsigned short* __restrict__ Bt,
                              float* __restrict__ C,
                              const float* __restrict__ bias,
                              int M, int N, int K, float alpha)
{
    __shared__ unsigned short sA[2][128 * 32];   // [row][k] bf16, 8 KB x2
    __shared__ unsigned short sB[2][128 * 32];   // [n][k]  bf16, 8 KB x2

    const int tid  = threadIdx.x;
    const int lane = tid & 31;
    const int wv   = tid >> 5;
    const int wm   = wv >> 1;        // 0..3 : 32-row slice
    const int wn   = wv & 1;         // 0..1 : 64-col slice
    const int r16  = lane & 15;
    const int hf   = lane >> 4;

    const int m0 = blockIdx.y * 128;
    const int n0 = blockIdx.x * 128;

    // Per-tile staging: 512+512 16B packets; each thread moves 2 A + 2 B.
    const int sr = tid >> 2;         // 0..63 row base
    const int sc = tid & 3;          // 16B packet within 64B row
    auto stage = [&](int kc, int buf) {
#pragma unroll
        for (int i = 0; i < 2; ++i) {
            int r = sr + i * 64;
            async_cp16(A  + (size_t)(m0 + r) * K + kc + sc * 8, &sA[buf][r * 32 + sc * 8]);
            async_cp16(Bt + (size_t)(n0 + r) * K + kc + sc * 8, &sB[buf][r * 32 + sc * 8]);
        }
    };

    floatx8 acc[2][4];
#pragma unroll
    for (int i = 0; i < 2; ++i)
#pragma unroll
        for (int j = 0; j < 4; ++j)
#pragma unroll
            for (int e = 0; e < 8; ++e)
                acc[i][j][e] = 0.0f;

    const int nk = K >> 5;
    stage(0, 0);
    int cur = 0;

    for (int it = 0; it < nk; ++it) {
        if (it + 1 < nk) {
            stage((it + 1) << 5, cur ^ 1);   // prefetch next tile, other buffer
            WAIT_ASYNC(4);                   // retire current tile (in-order)
        } else {
            WAIT_ASYNC(0);
        }
        __syncthreads();

        // A fragments (16x32 bf16 per ISA layout):
        // lanes 0-15: VGPR j<4 -> K=2j,2j+1 ; j>=4 -> K=16+2(j-4); lanes 16-31: +8.
        FragBF af[2];
#pragma unroll
        for (int tm = 0; tm < 2; ++tm) {
            int r = 32 * wm + 16 * tm + r16;
#pragma unroll
            for (int j = 0; j < 8; ++j) {
                int k0 = (j < 4) ? (8 * hf + 2 * j) : (16 + 8 * hf + 2 * (j - 4));
                af[tm].u[j] = *(const unsigned int*)&sA[cur][r * 32 + k0];
            }
        }
        // B fragments (32x16): same K order as A;
        // VGPR j<4 -> K=8*hf+2j,+1 ; j>=4 -> K=16+8*hf+2(j-4),+1 for column N = lane%16.
        FragBF bfrag[4];
#pragma unroll
        for (int tn = 0; tn < 4; ++tn) {
            int n = 64 * wn + 16 * tn + r16;
#pragma unroll
            for (int j = 0; j < 8; ++j) {
                int kb = (j < 4) ? (8 * hf + 2 * j) : (16 + 8 * hf + 2 * (j - 4));
                bfrag[tn].u[j] = *(const unsigned int*)&sB[cur][n * 32 + kb];
            }
        }
#pragma unroll
        for (int tm = 0; tm < 2; ++tm)
#pragma unroll
            for (int tn = 0; tn < 4; ++tn)
                acc[tm][tn] = __builtin_amdgcn_wmma_f32_16x16x32_bf16(
                    false, af[tm].v, false, bfrag[tn].v,
                    (short)0, acc[tm][tn], false, false);
        __syncthreads();     // all reads of buf[cur] done before it is restaged
        cur ^= 1;
    }

    // Epilogue: C layout (VGPR j: M = j + 8*hf, N = lane%16)
#pragma unroll
    for (int tm = 0; tm < 2; ++tm)
#pragma unroll
        for (int tn = 0; tn < 4; ++tn) {
            int n = n0 + 64 * wn + 16 * tn + r16;
            float bv = bias ? bias[n] : 0.0f;
#pragma unroll
            for (int j = 0; j < 8; ++j) {
                int m = m0 + 32 * wm + 16 * tm + 8 * hf + j;
                C[(size_t)m * N + n] = alpha * acc[tm][tn][j] + bv;
            }
        }
}

// ---------------------------------------------------------------------------
// Narrow GEMM: C(Mx16) = A(MxK) @ B(Kx16), f32 inputs (bottleneck proj).
// ---------------------------------------------------------------------------
__launch_bounds__(256)
__global__ void gemm_bf16_n16(const float* __restrict__ A,
                              const float* __restrict__ B,
                              float* __restrict__ C,
                              int M, int K)
{
    __shared__ unsigned int   sA[128 * 16];
    __shared__ unsigned short sB[16 * 32];

    const int tid  = threadIdx.x;
    const int lane = tid & 31;
    const int wv   = tid >> 5;
    const int r16  = lane & 15;
    const int hf   = lane >> 4;
    const int m0   = blockIdx.x * 128;

    floatx8 acc;
#pragma unroll
    for (int e = 0; e < 8; ++e) acc[e] = 0.0f;

    for (int kc = 0; kc < K; kc += 32) {
        const float* Ab = A + (size_t)m0 * K + kc;
#pragma unroll
        for (int i = 0; i < 4; ++i) {
            int s  = tid + i * 256;
            int r  = s >> 3;
            int c4 = s & 7;
            float4 f = *(const float4*)(Ab + (size_t)r * K + c4 * 4);
            sA[r * 16 + c4 * 2 + 0] = pack2bf(f.x, f.y);
            sA[r * 16 + c4 * 2 + 1] = pack2bf(f.z, f.w);
        }
        if (tid < 128) {
            int kr = tid >> 2;
            int c4 = tid & 3;
            float4 f = *(const float4*)(B + (size_t)(kc + kr) * 16 + c4 * 4);
            int nb = c4 * 4;
            sB[(nb + 0) * 32 + kr] = f2bf(f.x);
            sB[(nb + 1) * 32 + kr] = f2bf(f.y);
            sB[(nb + 2) * 32 + kr] = f2bf(f.z);
            sB[(nb + 3) * 32 + kr] = f2bf(f.w);
        }
        __syncthreads();

        FragBF af, bfrag;
        int r = 16 * wv + r16;
#pragma unroll
        for (int j = 0; j < 8; ++j) {
            int k0 = (j < 4) ? (8 * hf + 2 * j) : (16 + 8 * hf + 2 * (j - 4));
            af.u[j] = sA[r * 16 + (k0 >> 1)];
        }
#pragma unroll
        for (int j = 0; j < 8; ++j) {
            int kb = (j < 4) ? (8 * hf + 2 * j) : (16 + 8 * hf + 2 * (j - 4));
            bfrag.u[j] = *(const unsigned int*)&sB[r16 * 32 + kb];
        }
        acc = __builtin_amdgcn_wmma_f32_16x16x32_bf16(
            false, af.v, false, bfrag.v, (short)0, acc, false, false);
        __syncthreads();
    }
#pragma unroll
    for (int j = 0; j < 8; ++j) {
        int m = m0 + 16 * wv + 8 * hf + j;
        C[(size_t)m * 16 + r16] = acc[j];
    }
}

// ---------------------------------------------------------------------------
// decay[m][n] = exp( log_sigmoid( S1[m,:16] @ W2[:,n] + b2[n] ) / 16 )
// ---------------------------------------------------------------------------
__launch_bounds__(256)
__global__ void kg2_decay(const float* __restrict__ S1, const float* __restrict__ W2,
                          const float* __restrict__ b2, float* __restrict__ out,
                          int M, int N)
{
    int idx = blockIdx.x * 256 + threadIdx.x;
    if (idx >= M * N) return;
    int m = idx / N;
    int n = idx - m * N;
    const float* s = S1 + (size_t)m * 16;
    float acc = b2[n];
#pragma unroll
    for (int kk = 0; kk < 16; ++kk)
        acc = fmaf(s[kk], W2[kk * N + n], acc);
    float ls = (acc >= 0.0f) ? (-log1pf(__expf(-acc)))
                             : (acc - log1pf(__expf(acc)));
    out[idx] = __expf(ls * (1.0f / 16.0f));
}

// ---------------------------------------------------------------------------
// GLA scan. Grid = B*H*4: each block owns (b,h) and a 32-wide DV slice of
// the 64x128 state. 256 threads = kgroup(0..7) x vlocal(0..31), 8 state
// elements in registers. Writes o (pre-norm) and final state Sf.
// ---------------------------------------------------------------------------
__launch_bounds__(256)
__global__ void gla_scan(const float* __restrict__ q, const float* __restrict__ k,
                         const float* __restrict__ dc, const float* __restrict__ v,
                         float* __restrict__ o, float* __restrict__ Sf, int L)
{
    __shared__ float lq[64], lk[64], ld[64], lv[32], lp[256];

    const int bh = blockIdx.x >> 2;
    const int vs = blockIdx.x & 3;
    const int b  = bh >> 4;
    const int h  = bh & 15;
    const int tid = threadIdx.x;
    const int vl  = tid & 31;
    const int kg  = tid >> 5;
    const int kb  = kg * 8;
    const int vbase = vs * 32;

    float s[8];
#pragma unroll
    for (int i = 0; i < 8; ++i) s[i] = 0.0f;

    const size_t bq = ((size_t)b * L) * 1024 + (size_t)h * 64;
    const size_t bv = ((size_t)b * L) * 2048 + (size_t)h * 128 + vbase;

    for (int t = 0; t < L; ++t) {
        const size_t rq = bq + (size_t)t * 1024;
        if (tid < 64) {
            lq[tid] = q[rq + tid];
            lk[tid] = k[rq + tid];
            ld[tid] = dc[rq + tid];
        } else if (tid < 96) {
            lv[tid - 64] = v[bv + (size_t)t * 2048 + (tid - 64)];
        }
        __syncthreads();

        float vv  = lv[vl];
        float acc = 0.0f;
#pragma unroll
        for (int i = 0; i < 8; ++i) {
            s[i] = fmaf(ld[kb + i], s[i], lk[kb + i] * vv);
            acc  = fmaf(lq[kb + i], s[i], acc);
        }
        lp[tid] = acc;
        __syncthreads();

        if (tid < 32) {
            float r = lp[vl] + lp[32 + vl] + lp[64 + vl] + lp[96 + vl]
                    + lp[128 + vl] + lp[160 + vl] + lp[192 + vl] + lp[224 + vl];
            o[bv + (size_t)t * 2048 + vl] = r;
        }
    }

    const size_t sb = (size_t)bh * (64 * 128) + vbase;
#pragma unroll
    for (int i = 0; i < 8; ++i)
        Sf[sb + (size_t)(kb + i) * 128 + vl] = s[i];
}

// ---------------------------------------------------------------------------
// Per-row (128) LayerNorm (no affine) + SiLU(g) gating -> bf16 output.
// One wave per row; 8 rows per block.
// ---------------------------------------------------------------------------
__launch_bounds__(256)
__global__ void norm_gate_bf16(const float* __restrict__ o, const float* __restrict__ g,
                               unsigned short* __restrict__ y, int nrows)
{
    int row = blockIdx.x * 8 + (threadIdx.x >> 5);
    if (row >= nrows) return;
    int lane = threadIdx.x & 31;
    size_t off = (size_t)row * 128 + (size_t)lane * 4;

    float4 ov = *(const float4*)(o + off);
    float sum = ov.x + ov.y + ov.z + ov.w;
    float sq  = ov.x * ov.x + ov.y * ov.y + ov.z * ov.z + ov.w * ov.w;
#pragma unroll
    for (int m = 16; m >= 1; m >>= 1) {
        sum += __shfl_xor(sum, m, 32);
        sq  += __shfl_xor(sq,  m, 32);
    }
    float mean = sum * (1.0f / 128.0f);
    float var  = sq * (1.0f / 128.0f) - mean * mean;
    float rstd = rsqrtf(var + 1e-5f);

    float4 gv = *(const float4*)(g + off);
    float y0 = (ov.x - mean) * rstd * (gv.x / (1.0f + __expf(-gv.x)));
    float y1 = (ov.y - mean) * rstd * (gv.y / (1.0f + __expf(-gv.y)));
    float y2 = (ov.z - mean) * rstd * (gv.z / (1.0f + __expf(-gv.z)));
    float y3 = (ov.w - mean) * rstd * (gv.w / (1.0f + __expf(-gv.w)));
    uint2 u;
    u.x = pack2bf(y0, y1);
    u.y = pack2bf(y2, y3);
    *(uint2*)(y + off) = u;
}

// ---------------------------------------------------------------------------
// Host-side launch
// ---------------------------------------------------------------------------
extern "C" void kernel_launch(void* const* d_in, const int* in_sizes, int n_in,
                              void* d_out, int out_size, void* d_ws, size_t ws_size,
                              hipStream_t stream) {
    (void)in_sizes; (void)n_in; (void)out_size; (void)ws_size;

    const float* x    = (const float*)d_in[0];
    const float* Wq   = (const float*)d_in[1];
    const float* Wk   = (const float*)d_in[2];
    const float* Wkg1 = (const float*)d_in[3];
    const float* Wkg2 = (const float*)d_in[4];
    const float* bkg2 = (const float*)d_in[5];
    const float* Wv   = (const float*)d_in[6];
    const float* Wg   = (const float*)d_in[7];
    const float* bg   = (const float*)d_in[8];
    const float* Wo   = (const float*)d_in[9];

    const int Bn = 4, L = 2048, D = 2048, H = 16;
    const int M  = Bn * L;        // 8192
    const int Dh = D / 2;         // 1024
    const float SCALE = 0.08838834764831845f;  // 128^-0.5

    // workspace layout (~345 MB)
    char* w = (char*)d_ws;
    unsigned short* xb = (unsigned short*)w;  w += (size_t)M * D * 2;   // bf16 x; later reused as gated y
    unsigned short* Wt = (unsigned short*)w;  w += (size_t)D * D * 2;   // reused transposed weight
    float* q  = (float*)w;  w += (size_t)M * Dh * 4;
    float* kb = (float*)w;  w += (size_t)M * Dh * 4;
    float* dc = (float*)w;  w += (size_t)M * Dh * 4;
    float* s1 = (float*)w;  w += (size_t)M * 16 * 4;
    float* vb = (float*)w;  w += (size_t)M * D * 4;
    float* gb = (float*)w;  w += (size_t)M * D * 4;
    float* ob = (float*)w;  w += (size_t)M * D * 4;
    unsigned short* yb = xb;

    float* out = (float*)d_out;
    float* Sf  = out + (size_t)M * D;

    dim3 blk(256);

    conv_bf16<<<dim3((M * D / 4 + 255) / 256), blk, 0, stream>>>(x, xb, M * D / 4);

    convT_bf16<<<dim3(Dh / 32, D / 32), blk, 0, stream>>>(Wq, Wt, D, Dh);
    gemm_bf16bf16<<<dim3(Dh / 128, M / 128), blk, 0, stream>>>(xb, Wt, q,  nullptr, M, Dh, D, 1.0f);
    convT_bf16<<<dim3(Dh / 32, D / 32), blk, 0, stream>>>(Wk, Wt, D, Dh);
    gemm_bf16bf16<<<dim3(Dh / 128, M / 128), blk, 0, stream>>>(xb, Wt, kb, nullptr, M, Dh, D, SCALE);
    convT_bf16<<<dim3(D / 32, D / 32), blk, 0, stream>>>(Wv, Wt, D, D);
    gemm_bf16bf16<<<dim3(D / 128, M / 128), blk, 0, stream>>>(xb, Wt, vb, nullptr, M, D, D, 1.0f);
    convT_bf16<<<dim3(D / 32, D / 32), blk, 0, stream>>>(Wg, Wt, D, D);
    gemm_bf16bf16<<<dim3(D / 128, M / 128), blk, 0, stream>>>(xb, Wt, gb, bg, M, D, D, 1.0f);

    gemm_bf16_n16<<<dim3(M / 128), blk, 0, stream>>>(x, Wkg1, s1, M, D);
    kg2_decay<<<dim3((M * Dh + 255) / 256), blk, 0, stream>>>(s1, Wkg2, bkg2, dc, M, Dh);

    gla_scan<<<dim3(Bn * H * 4), blk, 0, stream>>>(q, kb, dc, vb, ob, Sf, L);
    norm_gate_bf16<<<dim3(M * H / 8), blk, 0, stream>>>(ob, gb, yb, M * H);

    convT_bf16<<<dim3(D / 32, D / 32), blk, 0, stream>>>(Wo, Wt, D, D);
    gemm_bf16bf16<<<dim3(D / 128, M / 128), blk, 0, stream>>>(yb, Wt, out, nullptr, M, D, D, 1.0f);
}
